// ManagedLSTM_24481313587731
// MI455X (gfx1250) — matrix operand order repaired and checked by reading, not compiled, not run on hardware
//
#include <hip/hip_runtime.h>

#define B_    64
#define T_    1024
#define IN_   256
#define H_    512
#define KTOT  768          // IN_ + H_
#define G4    2048         // 4 * H_

typedef unsigned int   u32;
typedef unsigned short u16;
typedef __attribute__((ext_vector_type(16))) __bf16 v16bf;
typedef __attribute__((ext_vector_type(8)))  float  v8f;
typedef __attribute__((ext_vector_type(4)))  u32    u32x4;

union Frag {
    v16bf bf;
    u32x4 q[2];
};

__device__ __forceinline__ u16 f2bf(float f) {
    u32 u = __float_as_uint(f);
    u32 r = u + 0x7FFFu + ((u >> 16) & 1u);   // round-to-nearest-even
    return (u16)(r >> 16);
}

__device__ __forceinline__ float sigf(float x) {
    return 1.0f / (1.0f + __expf(-x));
}

__device__ __forceinline__ v8f wmma_bf16(const Frag& a, const Frag& b, v8f c) {
    return __builtin_amdgcn_wmma_f32_16x16x32_bf16(
        /*neg_a=*/false, a.bf, /*neg_b=*/false, b.bf,
        /*c_mod=*/(short)0, c, /*reuse_a=*/false, /*reuse_b=*/false);
}

// Software grid barrier across the 16 co-resident workgroups of the
// persistent LSTM kernel. Release writes -> counter add -> spin (s_sleep
// backoff) -> acquire. One barrier per timestep (h is ping-pong buffered).
__device__ __forceinline__ void grid_barrier(unsigned* cnt, unsigned target) {
    __threadfence();                       // release this thread's writes
    __syncthreads();
    if (threadIdx.x == 0) {
        __hip_atomic_fetch_add(cnt, 1u, __ATOMIC_ACQ_REL,
                               __HIP_MEMORY_SCOPE_AGENT);
        while (__hip_atomic_load(cnt, __ATOMIC_ACQUIRE,
                                 __HIP_MEMORY_SCOPE_AGENT) < target) {
            __builtin_amdgcn_s_sleep(1);
        }
    }
    __syncthreads();
    __threadfence();                       // acquire other WGs' writes
}

// ---------------- prep kernels ----------------

// w_lstm [768][2048] f32 -> wT bf16 [2048][768]  (K contiguous for B-frags)
__global__ void k_cvt_wlstm(const float* __restrict__ w, u16* __restrict__ wT) {
    int i = blockIdx.x * blockDim.x + threadIdx.x;
    if (i >= KTOT * G4) return;
    int k = i / G4, n = i % G4;
    wT[(size_t)n * KTOT + k] = f2bf(w[i]);
}

// w_lin [512][512] f32 -> wlinT bf16 [512][512] transposed
__global__ void k_cvt_wlin(const float* __restrict__ w, u16* __restrict__ wT) {
    int i = blockIdx.x * blockDim.x + threadIdx.x;
    if (i >= H_ * H_) return;
    int k = i / H_, n = i % H_;
    wT[(size_t)n * H_ + k] = f2bf(w[i]);
}

__global__ void k_cvt_x(const float* __restrict__ x, u16* __restrict__ xb, int n) {
    int i = blockIdx.x * blockDim.x + threadIdx.x;
    if (i < n) xb[i] = f2bf(x[i]);
}

__global__ void k_zero(unsigned* __restrict__ p) { *p = 0u; }

// ---------------- persistent recurrent kernel ----------------
// All 1024 timesteps in ONE launch. Grid: 16 WGs x 256 threads (all
// co-resident). Each WG owns 32 hidden cols; each wave owns a (16-row,
// 16-hidden-col) tile with ALL FOUR gate tiles in registers, so the LSTM
// cell update needs no LDS and no cross-lane traffic. Cell state c (and
// final h) live entirely in registers across all timesteps.
//
// NOTE on the offset launder: the weight loads are loop-invariant across t;
// without defeating LICM the compiler hoists ~768 VGPRs of B-fragments out
// of the time loop and spills them to scratch (seen in round 2). Laundering
// the POINTER through asm kills its address-space provenance and demotes the
// loads to flat_load (seen in round 3). Laundering a zero OFFSET keeps the
// kernarg provenance (global_load_b128) while making addresses opaque per
// iteration so the loads stay inside the loop.
__global__ __launch_bounds__(256)
void k_lstm_all(const u16* __restrict__ xb, const u16* wTarg,
                const float* __restrict__ b_lstm,
                const float* __restrict__ h0, const float* __restrict__ c0,
                u16* __restrict__ hb0, u16* __restrict__ hb1,
                u16* __restrict__ hHist, float* __restrict__ outHC,
                unsigned* __restrict__ bar) {
    const int tid    = threadIdx.x;
    const int wave   = tid >> 5;
    const int lane   = tid & 31;
    const int laneLo = lane & 15;
    const int laneHi = lane >> 4;
    const int rowTile = wave & 3;                       // 4 row tiles (M = 64)
    const int colSub  = wave >> 2;                      // 2 col tiles per WG
    const int n0      = blockIdx.x * 32 + colSub * 16;  // hidden col base
    const int brow    = rowTile * 16 + laneLo;          // batch row for A frag
    const int ncol    = n0 + laneLo;                    // owned hidden column

    // Per-thread owned cells: rows rowTile*16 + e + 8*laneHi, column ncol.
    float creg[8], hreg[8];
#pragma unroll
    for (int e = 0; e < 8; ++e) {
        const int br  = rowTile * 16 + e + 8 * laneHi;
        const int idx = br * H_ + ncol;
        creg[e]  = c0[idx];
        hreg[e]  = h0[idx];
        hb0[idx] = f2bf(h0[idx]);
    }
    const float bi  = b_lstm[ncol];
    const float bg  = b_lstm[H_ + ncol];
    const float bff = b_lstm[2 * H_ + ncol] + 1.0f;     // haiku forget bias
    const float bo  = b_lstm[3 * H_ + ncol];

    unsigned target = 16u;
    grid_barrier(bar, target);                          // hb0 fully initialized
    target += 16u;

    const u16* xrow0 = xb + (size_t)brow * (T_ * IN_);

    for (int t = 0; t < T_; ++t) {
        const u16* hin  = (t & 1) ? hb1 : hb0;
        u16*       hout = (t & 1) ? hb0 : hb1;
        const u16* xrow = xrow0 + (size_t)t * IN_;
        const u16* hrow = hin + (size_t)brow * H_;

        // Opaque zero offset, renewed every timestep (see note above).
        size_t woff = 0;
        asm volatile("" : "+s"(woff));
        const u16* wT = wTarg + woff;

        v8f acc0 = {}, acc1 = {}, acc2 = {}, acc3 = {};

        for (int k0 = 0; k0 < KTOT; k0 += 32) {
            Frag a;
            if (k0 < IN_) {
                const u16* p = xrow + k0 + laneHi * 16;
                a.q[0] = *(const u32x4*)(p);
                a.q[1] = *(const u32x4*)(p + 8);
            } else {
                const u16* p = hrow + (k0 - IN_) + laneHi * 16;
                a.q[0] = *(const u32x4*)(p);
                a.q[1] = *(const u32x4*)(p + 8);
            }
            const int kb = k0 + laneHi * 16;
            const u16* wp = wT + (size_t)(n0 + laneLo) * KTOT + kb;
            Frag b0, b1, b2, b3;
            b0.q[0] = *(const u32x4*)(wp);
            b0.q[1] = *(const u32x4*)(wp + 8);
            wp += (size_t)H_ * KTOT;
            b1.q[0] = *(const u32x4*)(wp);
            b1.q[1] = *(const u32x4*)(wp + 8);
            wp += (size_t)H_ * KTOT;
            b2.q[0] = *(const u32x4*)(wp);
            b2.q[1] = *(const u32x4*)(wp + 8);
            wp += (size_t)H_ * KTOT;
            b3.q[0] = *(const u32x4*)(wp);
            b3.q[1] = *(const u32x4*)(wp + 8);

            acc0 = wmma_bf16(a, b0, acc0);   // gate i
            acc1 = wmma_bf16(a, b1, acc1);   // gate g
            acc2 = wmma_bf16(a, b2, acc2);   // gate f
            acc3 = wmma_bf16(a, b3, acc3);   // gate o
        }

        // elementwise LSTM cell update entirely from this wave's registers
#pragma unroll
        for (int e = 0; e < 8; ++e) {
            const int br  = rowTile * 16 + e + 8 * laneHi;
            const int idx = br * H_ + ncol;
            float iv = acc0[e] + bi;
            float gv = acc1[e] + bg;
            float fv = acc2[e] + bff;
            float ov = acc3[e] + bo;
            float cn = sigf(fv) * creg[e] + sigf(iv) * tanhf(gv);
            float hn = sigf(ov) * tanhf(cn);
            creg[e] = cn;
            hreg[e] = hn;
            u16 hb = f2bf(hn);
            hout[idx] = hb;
            hHist[(size_t)t * (B_ * H_) + idx] = hb;
        }

        if (t + 1 < T_) {                 // last step synced by kernel boundary
            grid_barrier(bar, target);
            target += 16u;
        }
    }

    // finals straight into d_out: h_fin then c_fin, each [B, H]
#pragma unroll
    for (int e = 0; e < 8; ++e) {
        const int br  = rowTile * 16 + e + 8 * laneHi;
        const int idx = br * H_ + ncol;
        outHC[idx]           = hreg[e];
        outHC[B_ * H_ + idx] = creg[e];
    }
}

// ---------------- output projection ----------------
// y = hHist[T*B, 512] @ wlinT + b_lin, one big WMMA GEMM.
// Grid (1024, 8) x 256 threads; WG tile = 64 rows x 64 cols; wave = 1 row tile
// x 2 col tiles.
__global__ __launch_bounds__(256)
void k_out_gemm(const u16* __restrict__ hHist, const u16* __restrict__ wlinT,
                const float* __restrict__ b_lin, float* __restrict__ y) {
    const int tid    = threadIdx.x;
    const int wave   = tid >> 5;
    const int lane   = tid & 31;
    const int laneLo = lane & 15;
    const int laneHi = lane >> 4;
    const int rowTile = wave & 3;
    const int colPair = wave >> 2;
    const int R   = blockIdx.x * 64 + rowTile * 16 + laneLo;   // row in [0, T*B)
    const int nb0 = blockIdx.y * 64 + colPair * 32;            // 2 x 16 cols

    const u16* arow = hHist + (size_t)R * H_;

    v8f acc0 = {}, acc1 = {};
    for (int k0 = 0; k0 < H_; k0 += 32) {
        Frag a;
        const u16* p = arow + k0 + laneHi * 16;
        a.q[0] = *(const u32x4*)(p);
        a.q[1] = *(const u32x4*)(p + 8);
        const int kb = k0 + laneHi * 16;
        Frag b0, b1;
        const u16* w0 = wlinT + (size_t)(nb0 + laneLo) * H_ + kb;
        b0.q[0] = *(const u32x4*)(w0);
        b0.q[1] = *(const u32x4*)(w0 + 8);
        const u16* w1 = w0 + (size_t)16 * H_;
        b1.q[0] = *(const u32x4*)(w1);
        b1.q[1] = *(const u32x4*)(w1 + 8);
        acc0 = wmma_bf16(a, b0, acc0);
        acc1 = wmma_bf16(a, b1, acc1);
    }

#pragma unroll
    for (int e = 0; e < 8; ++e) {
        const int rowIdx = blockIdx.x * 64 + rowTile * 16 + e + 8 * laneHi;
        const int tt = rowIdx >> 6;      // hHist rows are [t][b], B_=64
        const int bb = rowIdx & 63;
        const size_t base = (size_t)bb * (T_ * 512) + (size_t)tt * 512;
        const int n0c = nb0 + laneLo;
        const int n1c = n0c + 16;
        y[base + n0c] = acc0[e] + b_lin[n0c];
        y[base + n1c] = acc1[e] + b_lin[n1c];
    }
}

// ---------------- launch ----------------
extern "C" void kernel_launch(void* const* d_in, const int* in_sizes, int n_in,
                              void* d_out, int out_size, void* d_ws, size_t ws_size,
                              hipStream_t stream) {
    const float* x      = (const float*)d_in[0];
    const float* h0     = (const float*)d_in[1];
    const float* c0     = (const float*)d_in[2];
    const float* w_lstm = (const float*)d_in[3];
    const float* b_lstm = (const float*)d_in[4];
    const float* w_lin  = (const float*)d_in[5];
    const float* b_lin  = (const float*)d_in[6];
    float* y = (float*)d_out;

    char* ws = (char*)d_ws;
    size_t off = 0;
    auto alloc = [&](size_t bytes) -> char* {
        char* p = ws + off;
        off = (off + bytes + 255) & ~(size_t)255;
        return p;
    };

    u16*      wT    = (u16*)alloc((size_t)KTOT * G4 * 2);     // 3 MB (L2-resident)
    u16*      wlinT = (u16*)alloc((size_t)H_ * H_ * 2);       // 0.5 MB
    u16*      xbf   = (u16*)alloc((size_t)B_ * T_ * IN_ * 2); // 32 MB
    u16*      hb0   = (u16*)alloc((size_t)B_ * H_ * 2);       // ping
    u16*      hb1   = (u16*)alloc((size_t)B_ * H_ * 2);       // pong
    u16*      hHist = (u16*)alloc((size_t)T_ * B_ * H_ * 2);  // 64 MB
    unsigned* bar   = (unsigned*)alloc(sizeof(unsigned));

    k_cvt_wlstm<<<(KTOT * G4 + 255) / 256, 256, 0, stream>>>(w_lstm, wT);
    k_cvt_wlin<<<(H_ * H_ + 255) / 256, 256, 0, stream>>>(w_lin, wlinT);
    {
        int n = B_ * T_ * IN_;
        k_cvt_x<<<(n + 255) / 256, 256, 0, stream>>>(x, xbf, n);
    }
    k_zero<<<1, 1, 0, stream>>>(bar);

    // one persistent launch for all 1024 timesteps
    k_lstm_all<<<16, 256, 0, stream>>>(xbf, wT, b_lstm, h0, c0, hb0, hb1,
                                       hHist, y + (size_t)B_ * T_ * 512, bar);

    k_out_gemm<<<dim3(1024, 8), 256, 0, stream>>>(hHist, wlinT, b_lin, y);
}
